// TripletLoss_68212670595762
// MI455X (gfx1250) — matrix operand order repaired and checked by reading, not compiled, not run on hardware
//
#include <hip/hip_runtime.h>

// ---------------------------------------------------------------------------
// TripletMarginLoss (cosine, easy-positive / semihard-negative) for gfx1250.
// sim = normalize(E) @ normalize(E)^T computed with v_wmma_f32_16x16x32_f16.
// N = 16384 anchors, D = 128, margin = 0.2, AvgNonZeroReducer.
// ---------------------------------------------------------------------------

typedef _Float16 v4h  __attribute__((ext_vector_type(4)));
typedef _Float16 v8h  __attribute__((ext_vector_type(8)));
typedef _Float16 v16h __attribute__((ext_vector_type(16)));
typedef float    v8f  __attribute__((ext_vector_type(8)));

#define NROW 16384
#define DIM  128
#define TMARGIN 0.2f
#define SENT (-1e9f)

// Order-preserving float <-> uint32 map so we can use deterministic atomicMax.
__device__ __forceinline__ unsigned fkey(float f) {
    unsigned u = __float_as_uint(f);
    return (u & 0x80000000u) ? ~u : (u | 0x80000000u);
}
__device__ __forceinline__ float funkey(unsigned k) {
    unsigned u = (k & 0x80000000u) ? (k ^ 0x80000000u) : ~k;
    return __uint_as_float(u);
}

// ---------------------------------------------------------------------------
// Kernel 1: L2-normalize rows (f32 math), emit row-major f16 matrix, and
// initialize the positive/negative running-max keys to the sentinel.
// One wave32 per row; each lane owns 4 of the 128 elements.
// ---------------------------------------------------------------------------
__global__ void k_normalize(const float* __restrict__ emb,
                            _Float16* __restrict__ eh,
                            unsigned* __restrict__ pkey,
                            unsigned* __restrict__ nkey) {
    const int row  = blockIdx.x * 8 + (threadIdx.x >> 5);
    const int lane = threadIdx.x & 31;
    const float4 v = *(const float4*)(emb + (size_t)row * DIM + lane * 4);
    float s = v.x * v.x + v.y * v.y + v.z * v.z + v.w * v.w;
    #pragma unroll
    for (int m = 16; m >= 1; m >>= 1) s += __shfl_xor(s, m, 32);
    const float inv = 1.0f / sqrtf(s);
    v4h h;
    h[0] = (_Float16)(v.x * inv);
    h[1] = (_Float16)(v.y * inv);
    h[2] = (_Float16)(v.z * inv);
    h[3] = (_Float16)(v.w * inv);
    *(v4h*)(eh + (size_t)row * DIM + lane * 4) = h;
    if (lane == 0) {
        pkey[row] = fkey(SENT);
        nkey[row] = fkey(SENT);
    }
}

// ---------------------------------------------------------------------------
// Kernels 2/3: masked row-max over the similarity matrix.
//   PASS2 == 0 : positives   mask = (y_i == y_j) && (i != j)   -> pkey
//   PASS2 == 1 : semihard    mask = (y_i != y_j) && (s < p_i)  -> nkey
// Grid: one block per 16-row i-tile (1024 blocks); 8 waves per block each
// cover 128 of the 1024 j-tiles. A-tile is preloaded once per wave.
//
// VGPR layouts per CDNA5 ISA 7.12.2 (wave32):
//   A 16x32 f16 : lane%16 = row M; lanes<16 hold K 0-7 & 16-23, lanes>=16
//                 hold K 8-15 & 24-31 (per 32-wide K chunk).
//   B 32x16 f16 : lane%16 = col N; lanes<16 hold K 0-7 & 16-23, lanes>=16 K 8-15 & 24-31.
//   C/D 16x16 f32: lane%16 = col N; reg r = row M (lanes<16) / M=8+r (>=16).
// ---------------------------------------------------------------------------
template <int PASS2>
__global__ void k_pass(const _Float16* __restrict__ eh,
                       const int* __restrict__ y,
                       const unsigned* __restrict__ pkey_in,
                       unsigned* __restrict__ keyout) {
    const int itile = blockIdx.x;
    const int wave  = threadIdx.x >> 5;
    const int lane  = threadIdx.x & 31;
    const int half  = lane >> 4;
    const int l15   = lane & 15;
    const int ibase = itile * 16;

    // --- preload A tile (this lane's row, all 4 K-chunks) -------------------
    const _Float16* arow = eh + (size_t)(ibase + l15) * DIM;
    v16h A[4];
    #pragma unroll
    for (int c = 0; c < 4; ++c) {
        v8h lo = *(const v8h*)(arow + c * 32 + half * 8);        // K c*32 + {0-7 | 8-15}
        v8h hi = *(const v8h*)(arow + c * 32 + 16 + half * 8);   // K c*32 + {16-23 | 24-31}
        A[c] = __builtin_shufflevector(lo, hi, 0, 1, 2, 3, 4, 5, 6, 7,
                                               8, 9, 10, 11, 12, 13, 14, 15);
    }

    // --- per-C-row metadata (rows this lane accumulates) --------------------
    int   yi[8];
    float pthr[8];
    #pragma unroll
    for (int r = 0; r < 8; ++r) {
        const int m = ibase + half * 8 + r;
        yi[r] = y[m];
        if (PASS2) pthr[r] = funkey(pkey_in[m]);
        else       pthr[r] = 0.0f;
    }

    float rmax[8];
    #pragma unroll
    for (int r = 0; r < 8; ++r) rmax[r] = SENT;

    const int jt0 = wave * 128;
    const int jt1 = jt0 + 128;
    for (int jt = jt0; jt < jt1; ++jt) {
        const int jb   = jt * 16;
        const int jcol = jb + l15;             // this lane's C column
        const int yj   = y[jcol];
        const _Float16* brow = eh + (size_t)jcol * DIM + half * 8;

        v8f acc = {};
        #pragma unroll
        for (int c = 0; c < 4; ++c) {
            v8h lo = *(const v8h*)(brow + c * 32);
            v8h hi = *(const v8h*)(brow + c * 32 + 16);
            v16h B = __builtin_shufflevector(lo, hi, 0, 1, 2, 3, 4, 5, 6, 7,
                                                     8, 9, 10, 11, 12, 13, 14, 15);
            acc = __builtin_amdgcn_wmma_f32_16x16x32_f16(
                /*neg_a=*/false, A[c], /*neg_b=*/false, B,
                /*c_mod=*/(short)0, acc, /*reuse_a=*/false, /*reuse_b=*/false);
        }

        #pragma unroll
        for (int r = 0; r < 8; ++r) {
            const int   m = ibase + half * 8 + r;
            const float s = acc[r];
            bool cond;
            if (PASS2) cond = (yi[r] != yj) && (s < pthr[r]);
            else       cond = (yi[r] == yj) && (m != jcol);
            if (cond && s > rmax[r]) rmax[r] = s;
        }
    }

    // --- reduce row maxima across the 16 lanes of each half -----------------
    #pragma unroll
    for (int r = 0; r < 8; ++r) {
        float v = rmax[r];
        #pragma unroll
        for (int m = 8; m >= 1; m >>= 1) v = fmaxf(v, __shfl_xor(v, m, 16));
        rmax[r] = v;
    }
    if (l15 == 0) {
        #pragma unroll
        for (int r = 0; r < 8; ++r) {
            const int m = ibase + half * 8 + r;
            atomicMax(keyout + m, fkey(rmax[r]));
        }
    }
}

// ---------------------------------------------------------------------------
// Kernel 4: per-anchor hinge loss.
// ---------------------------------------------------------------------------
__global__ void k_loss(const unsigned* __restrict__ pkey,
                       const unsigned* __restrict__ nkey,
                       float* __restrict__ losses) {
    const int i = blockIdx.x * blockDim.x + threadIdx.x;
    if (i >= NROW) return;
    const float p = funkey(pkey[i]);
    const float n = funkey(nkey[i]);
    float loss = 0.0f;
    if (p > -1e8f && n > -1e8f) loss = fmaxf(0.0f, TMARGIN + n - p);
    losses[i] = loss;
}

// ---------------------------------------------------------------------------
// Kernel 5: deterministic single-block reduction: mean over nonzero losses.
// ---------------------------------------------------------------------------
__global__ void k_reduce(const float* __restrict__ losses,
                         float* __restrict__ out) {
    __shared__ float ssum[256];
    __shared__ int   scnt[256];
    float s = 0.0f;
    int   c = 0;
    for (int i = threadIdx.x; i < NROW; i += 256) {
        const float l = losses[i];
        s += l;
        c += (l > 0.0f) ? 1 : 0;
    }
    ssum[threadIdx.x] = s;
    scnt[threadIdx.x] = c;
    __syncthreads();
    for (int m = 128; m >= 1; m >>= 1) {
        if (threadIdx.x < m) {
            ssum[threadIdx.x] += ssum[threadIdx.x + m];
            scnt[threadIdx.x] += scnt[threadIdx.x + m];
        }
        __syncthreads();
    }
    if (threadIdx.x == 0) {
        const int nnz = scnt[0] > 1 ? scnt[0] : 1;
        out[0] = ssum[0] / (float)nnz;
    }
}

// ---------------------------------------------------------------------------
extern "C" void kernel_launch(void* const* d_in, const int* in_sizes, int n_in,
                              void* d_out, int out_size, void* d_ws, size_t ws_size,
                              hipStream_t stream) {
    const float* emb = (const float*)d_in[0];
    const int*   y   = (const int*)d_in[1];
    float*       out = (float*)d_out;

    // Workspace layout (needs ~4.4 MB):
    //   [0, 4MB)        : f16 normalized embeddings (16384 x 128)
    //   then 3 x 64 KB  : pkey (u32), nkey (u32), losses (f32)
    char* ws = (char*)d_ws;
    _Float16* eh    = (_Float16*)ws;
    unsigned* pkey  = (unsigned*)(ws + (size_t)NROW * DIM * sizeof(_Float16));
    unsigned* nkey  = pkey + NROW;
    float*    losses = (float*)(nkey + NROW);

    k_normalize<<<NROW / 8, 256, 0, stream>>>(emb, eh, pkey, nkey);
    k_pass<0><<<NROW / 16, 256, 0, stream>>>(eh, y, pkey, pkey);
    k_pass<1><<<NROW / 16, 256, 0, stream>>>(eh, y, pkey, nkey);
    k_loss<<<NROW / 256, 256, 0, stream>>>(pkey, nkey, losses);
    k_reduce<<<1, 256, 0, stream>>>(losses, out);
}
